// MHA_4956392259611
// MI455X (gfx1250) — matrix operand order repaired and checked by reading, not compiled, not run on hardware
//
#include <hip/hip_runtime.h>
#include <hip/hip_bf16.h>
#include <math.h>

#define BSZ  4
#define SLEN 2048
#define DIN  1024
#define EMB  1024
#define HNO  16
#define HSZ  64

typedef __attribute__((ext_vector_type(16))) __bf16        v16bf;
typedef __attribute__((ext_vector_type(8)))  float         v8f;
typedef __attribute__((ext_vector_type(8)))  unsigned int  v8u;

__device__ __forceinline__ unsigned short f2bf(float f) {
    unsigned int u = __builtin_bit_cast(unsigned int, f);
    u += 0x7fffu + ((u >> 16) & 1u);       // round-to-nearest-even
    return (unsigned short)(u >> 16);
}
__device__ __forceinline__ unsigned int pack_bf2(float lo, float hi) {
    return (unsigned int)f2bf(lo) | ((unsigned int)f2bf(hi) << 16);
}
__device__ __forceinline__ v16bf as_v16bf(v8u u) {
    return __builtin_bit_cast(v16bf, u);
}
__device__ __forceinline__ v8f wmma_bf16(v16bf a, v16bf b, v8f c) {
    return __builtin_amdgcn_wmma_f32_16x16x32_bf16(
        false, a, false, b, (short)0, c, false, false);
}

// gfx1250 async global->LDS copy engine (ASYNCcnt). Per ISA 10.2, a flat
// shared-memory address maps to LDS via addr[31:0], so truncating the generic
// pointer yields the LDS byte offset the VDST operand wants.
__device__ __forceinline__ void async_g2l_b128(const void* gptr, void* lptr) {
    asm volatile("global_load_async_to_lds_b128 %0, %1, off"
                 :
                 : "v"((unsigned)(size_t)lptr),
                   "v"((unsigned long long)(size_t)gptr)
                 : "memory");
}
__device__ __forceinline__ void wait_asynccnt0() {
    asm volatile("s_wait_asynccnt 0x0" ::: "memory");
}

// ---------------------------------------------------------------------------
// Kernel 1: fused QKV projection. y[m,e] = sum_d x[m,d]*W[e,d] + b[e]
// Output stored bf16 in [b, h, s, hd] layout for the attention kernel.
// Block: 256 threads (8 waves). Tile: M=128, N=64, K-step=32.
// blockIdx.z selects {Q,K,V}. (f32->bf16 conversion prevents async use here.)
// ---------------------------------------------------------------------------
__global__ __launch_bounds__(256) void qkv_proj_kernel(
    const float* __restrict__ x,
    const float* __restrict__ Wq, const float* __restrict__ bq,
    const float* __restrict__ Wk, const float* __restrict__ bk,
    const float* __restrict__ Wv, const float* __restrict__ bv,
    unsigned short* __restrict__ qb,
    unsigned short* __restrict__ kb,
    unsigned short* __restrict__ vb)
{
    __shared__ unsigned int a_u[128 * 16];   // 128 rows x 32 k  (bf16 pairs)
    __shared__ unsigned int b_u[64 * 16];    // 64 e-rows x 32 k (bf16 pairs)

    const int tid  = threadIdx.x;
    const int wave = tid >> 5;
    const int lane = tid & 31;
    const int half = lane >> 4;
    const int l16  = lane & 15;

    const int mbase = blockIdx.x * 128;
    const int nbase = blockIdx.y * 64;
    const int z     = blockIdx.z;

    const float* W    = (z == 0) ? Wq : (z == 1) ? Wk : Wv;
    const float* bias = (z == 0) ? bq : (z == 1) ? bk : bv;
    unsigned short* dst = (z == 0) ? qb : (z == 1) ? kb : vb;

    v8f c[4] = {};

    for (int k0 = 0; k0 < DIN; k0 += 32) {
        __syncthreads();
        // A tile: x[mbase..+128, k0..+32] -> bf16 pairs
        for (int p = tid; p < 128 * 16; p += 256) {
            int m = p >> 4, kp = p & 15;
            const float* src = x + (size_t)(mbase + m) * DIN + k0 + kp * 2;
            a_u[p] = pack_bf2(src[0], src[1]);
        }
        // B tile: W[nbase..+64, k0..+32] -> bf16 pairs (B[k][n] = W[n][k])
        for (int p = tid; p < 64 * 16; p += 256) {
            int e = p >> 4, kp = p & 15;
            const float* src = W + (size_t)(nbase + e) * DIN + k0 + kp * 2;
            b_u[p] = pack_bf2(src[0], src[1]);
        }
        __syncthreads();

        // A-fragment: lane m = l16, vgpr i holds K pair per ISA 16-bit A layout
        v8u au;
#pragma unroll
        for (int i = 0; i < 8; ++i) {
            int k = ((i >> 2) << 4) + (half << 3) + ((i & 3) << 1);
            au[i] = a_u[(wave * 16 + l16) * 16 + (k >> 1)];
        }
        v16bf a = as_v16bf(au);

#pragma unroll
        for (int t = 0; t < 4; ++t) {
            v8u bu;
#pragma unroll
            for (int i = 0; i < 8; ++i)
                bu[i] = b_u[(t * 16 + l16) * 16 + ((i >> 2) << 3) + half * 4 + (i & 3)]; // k = 16*(i>>2)+8*half+2*(i&3)
            c[t] = wmma_bf16(a, as_v16bf(bu), c[t]);
        }
    }

    // epilogue: add bias, convert bf16, scatter to [b,h,s,hd]
#pragma unroll
    for (int t = 0; t < 4; ++t) {
        int e = nbase + t * 16 + l16;
        float be = bias[e];
        int h = e >> 6, hd = e & 63;
#pragma unroll
        for (int v = 0; v < 8; ++v) {
            int mg = mbase + wave * 16 + half * 8 + v;
            int bi = mg >> 11, s = mg & 2047;
            dst[((size_t)(bi * HNO + h) * SLEN + s) * HSZ + hd] =
                f2bf(c[t][v] + be);
        }
    }
}

// ---------------------------------------------------------------------------
// Kernel 2: causal flash attention per (b, h, 128-query strip).
// 8 waves; wave owns a 16-row query strip. Online softmax kept in registers
// (row r = half*8+v lives in VGPR v of the C fragments, matching m/l lanes).
// Q strip and K tiles staged via the async global->LDS engine.
// Output written directly in the reference's [b,h,hd,s] flattening.
// ---------------------------------------------------------------------------
__global__ __launch_bounds__(256) void flash_attn_kernel(
    const unsigned short* __restrict__ qb,
    const unsigned short* __restrict__ kb,
    const unsigned short* __restrict__ vb,
    float* __restrict__ out)
{
    __shared__ unsigned int q_u[128 * 32];   // 128 q-rows x 64 hd (bf16 pairs)
    __shared__ unsigned int k_u[64 * 32];    // 64 keys  x 64 hd
    __shared__ unsigned int vt_u[64 * 32];   // 64 hd    x 64 keys (transposed)
    __shared__ unsigned int p_u[8][16 * 32]; // per-wave P tile 16 x 64 bf16

    const int tid  = threadIdx.x;
    const int wave = tid >> 5;
    const int lane = tid & 31;
    const int half = lane >> 4;
    const int l16  = lane & 15;

    const int qbase = blockIdx.x * 128;
    const int h     = blockIdx.y;
    const int b     = blockIdx.z;
    const size_t bh = (size_t)(b * HNO + h) * SLEN;

    // stage Q strip: 16 KB straight bf16 copy via async engine
    {
        const char* qsrc = (const char*)(qb + (bh + qbase) * HSZ);
        char*       qdst = (char*)q_u;
        for (int p = tid; p < 1024; p += 256)
            async_g2l_b128(qsrc + (size_t)p * 16, qdst + p * 16);
        wait_asynccnt0();
    }
    __syncthreads();

    // Q A-fragments for hd-chunks 0..31 and 32..63
    v16bf qa[2];
#pragma unroll
    for (int ch = 0; ch < 2; ++ch) {
        v8u au;
#pragma unroll
        for (int i = 0; i < 8; ++i) {
            int k = ch * 32 + ((i >> 2) << 4) + (half << 3) + ((i & 3) << 1);
            au[i] = q_u[(wave * 16 + l16) * 32 + (k >> 1)];
        }
        qa[ch] = as_v16bf(au);
    }

    v8f o[4] = {};
    float mrow[8], lrow[8];
#pragma unroll
    for (int v = 0; v < 8; ++v) { mrow[v] = -__builtin_inff(); lrow[v] = 0.f; }

    const int jend = qbase + 128;
    for (int j = 0; j < jend; j += 64) {
        __syncthreads();
        // K tile (8 KB, [key][hd]): async straight copy
        {
            const char* ksrc = (const char*)(kb + (bh + j) * HSZ);
            char*       kdst = (char*)k_u;
            for (int p = tid; p < 512; p += 256)
                async_g2l_b128(ksrc + (size_t)p * 16, kdst + p * 16);
        }
        // V tile: coalesced 8-byte loads, transposed b16 stores -> [hd][key]
        {
            const uint2* vsrc8 = (const uint2*)(vb + (bh + j) * HSZ);
            unsigned short* vt_s = (unsigned short*)vt_u;
            for (int p = tid; p < 1024; p += 256) {
                int key = p >> 4;           // 64 keys
                int hd4 = (p & 15) * 4;     // 16 chunks of 4 hd
                uint2 d = vsrc8[p];
                vt_s[(hd4 + 0) * 64 + key] = (unsigned short)(d.x & 0xffffu);
                vt_s[(hd4 + 1) * 64 + key] = (unsigned short)(d.x >> 16);
                vt_s[(hd4 + 2) * 64 + key] = (unsigned short)(d.y & 0xffffu);
                vt_s[(hd4 + 3) * 64 + key] = (unsigned short)(d.y >> 16);
            }
        }
        wait_asynccnt0();
        __syncthreads();

        // S = Q * K^T  (B[k=hd][n=key] pairs contiguous in k_u rows)
        v8f sfr[4];
#pragma unroll
        for (int t = 0; t < 4; ++t) {
            v8f cc = {};
#pragma unroll
            for (int ch = 0; ch < 2; ++ch) {
                v8u bu;
#pragma unroll
                for (int i = 0; i < 8; ++i) {
                    int k = ch * 32 + ((i >> 2) << 4) + (half << 3) + ((i & 3) << 1);
                    bu[i] = k_u[(t * 16 + l16) * 32 + (k >> 1)];
                }
                cc = wmma_bf16(qa[ch], as_v16bf(bu), cc);
            }
            sfr[t] = cc;
        }

        // scale, causal mask, online softmax (register resident)
#pragma unroll
        for (int v = 0; v < 8; ++v) {
            int row = qbase + wave * 16 + half * 8 + v;
            float mx = -__builtin_inff();
#pragma unroll
            for (int t = 0; t < 4; ++t) {
                int col = j + t * 16 + l16;
                float s = sfr[t][v] * 0.125f;           // 1/sqrt(64)
                s = (col <= row) ? s : -__builtin_inff();
                sfr[t][v] = s;
                mx = fmaxf(mx, s);
            }
#pragma unroll
            for (int mk = 1; mk < 16; mk <<= 1)
                mx = fmaxf(mx, __shfl_xor(mx, mk, 32));
            float mnew = fmaxf(mrow[v], mx);

            float sum = 0.f;
#pragma unroll
            for (int t = 0; t < 4; ++t) {
                float p = __expf(sfr[t][v] - mnew);
                sfr[t][v] = p;
                sum += p;
            }
#pragma unroll
            for (int mk = 1; mk < 16; mk <<= 1)
                sum += __shfl_xor(sum, mk, 32);

            float alpha = __expf(mrow[v] - mnew);
            lrow[v] = lrow[v] * alpha + sum;
            mrow[v] = mnew;
#pragma unroll
            for (int t = 0; t < 4; ++t) o[t][v] *= alpha;
        }

        // C-layout P -> bf16 A-layout via per-wave LDS buffer
        unsigned short* ps = (unsigned short*)p_u[wave];
#pragma unroll
        for (int t = 0; t < 4; ++t)
#pragma unroll
            for (int v = 0; v < 8; ++v)
                ps[(half * 8 + v) * 64 + t * 16 + l16] = f2bf(sfr[t][v]);
        __syncthreads();

        // O += P * V
        const unsigned int* pu = p_u[wave];
#pragma unroll
        for (int ch = 0; ch < 2; ++ch) {
            v8u au;
#pragma unroll
            for (int i = 0; i < 8; ++i) {
                int k = ch * 32 + ((i >> 2) << 4) + (half << 3) + ((i & 3) << 1);
                au[i] = pu[l16 * 32 + (k >> 1)];
            }
            v16bf pa = as_v16bf(au);
#pragma unroll
            for (int t = 0; t < 4; ++t) {
                v8u bu;
#pragma unroll
                for (int i = 0; i < 8; ++i) {
                    int k = ch * 32 + ((i >> 2) << 4) + (half << 3) + ((i & 3) << 1);
                    bu[i] = vt_u[(t * 16 + l16) * 32 + (k >> 1)];
                }
                o[t] = wmma_bf16(pa, as_v16bf(bu), o[t]);
            }
        }
    }

    // epilogue: normalize and store as out[b][h][hd][s] (reference flattening)
#pragma unroll
    for (int t = 0; t < 4; ++t) {
        int hd = t * 16 + l16;
#pragma unroll
        for (int v = 0; v < 8; ++v) {
            int s = qbase + wave * 16 + half * 8 + v;
            out[bh * HSZ + (size_t)hd * SLEN + s] = o[t][v] / lrow[v];
        }
    }
}

// ---------------------------------------------------------------------------
extern "C" void kernel_launch(void* const* d_in, const int* in_sizes, int n_in,
                              void* d_out, int out_size, void* d_ws, size_t ws_size,
                              hipStream_t stream) {
    const float* x  = (const float*)d_in[0];
    const float* Wq = (const float*)d_in[1];
    const float* bq = (const float*)d_in[2];
    const float* Wk = (const float*)d_in[3];
    const float* bk = (const float*)d_in[4];
    const float* Wv = (const float*)d_in[5];
    const float* bv = (const float*)d_in[6];

    // workspace: Q,K,V bf16 [b,h,s,hd] = 3 x 16 MB = 48 MB
    const size_t nqkv = (size_t)BSZ * SLEN * EMB;
    unsigned short* qb = (unsigned short*)d_ws;
    unsigned short* kb = qb + nqkv;
    unsigned short* vb = kb + nqkv;

    dim3 gproj((BSZ * SLEN) / 128, EMB / 64, 3);
    qkv_proj_kernel<<<gproj, 256, 0, stream>>>(x, Wq, bq, Wk, bk, Wv, bv,
                                               qb, kb, vb);

    dim3 gattn(SLEN / 128, HNO, BSZ);
    flash_attn_kernel<<<gattn, 256, 0, stream>>>(qb, kb, vb, (float*)d_out);
}
